// MultiHeadSpectralAttention_87746181857718
// MI455X (gfx1250) — matrix operand order repaired and checked by reading, not compiled, not run on hardware
//
#include <hip/hip_runtime.h>
#include <hip/hip_bf16.h>
#include <math.h>

// MI455X / gfx1250: wave32, WMMA bf16 16x16x32 path.
// Pipeline: cvt->bf16, build circulant filter w_ext = ifft(symmetrized mask),
// GEMM1 qkv, GEMM2 circulant filter (q,k), GEMM3 scores, softmax,
// GEMM4 attn@v, GEMM5 proj (+bias, f32 out).
// Each wave computes a 32x64 tile: 2 A-frags x 4 B-frags -> 8 WMMAs per K-step,
// with all fragment loads hoisted ahead of the WMMA burst.

typedef __attribute__((ext_vector_type(16))) __bf16 v16bf;
typedef __attribute__((ext_vector_type(8)))  __bf16 v8bf;
typedef __attribute__((ext_vector_type(8)))  float  v8f;

#define WMMA_BF16(a, b, c) \
  __builtin_amdgcn_wmma_f32_16x16x32_bf16(false, (a), false, (b), (short)0, (c), false, false)

namespace {
constexpr int BATCH = 2;
constexpr int SEQ   = 2048;
constexpr int DIM   = 768;
constexpr int HEADS = 12;
constexpr int HD    = 64;        // head dim
constexpr int QKV   = 3 * DIM;   // 2304
constexpr int BH    = BATCH * HEADS; // 24
}

// ---------- fragment loaders (CDNA5 16-bit WMMA layouts, wave32) ----------

// A: 16x32 (MxK), row-major source with leading dim lda (elements).
// lanes 0-15: M=lane, frag[0..7]=K 0..7, frag[8..15]=K 16..23
// lanes16-31: M=lane-16, frag[0..7]=K 8..15, frag[8..15]=K 24..31
__device__ __forceinline__ v16bf load_a_frag(const __bf16* __restrict__ A, int lda) {
  const int lane = threadIdx.x & 31;
  const int m    = lane & 15;
  const int koff = (lane >> 4) << 3;   // 0 or 8
  const __bf16* p = A + (size_t)m * lda + koff;
  v8bf lo = *(const v8bf*)p;
  v8bf hi = *(const v8bf*)(p + 16);
  v16bf f;
#pragma unroll
  for (int j = 0; j < 8; ++j) { f[j] = lo[j]; f[8 + j] = hi[j]; }
  return f;
}

// B: 32x16 (KxN). Source stored "NT" style: Bw[col][k], K contiguous, ld = ldb.
// lanes 0-15: N=lane, frag[0..7]=K 0..7, frag[8..15]=K 16..23
// lanes16-31: N=lane-16, frag[0..7]=K 8..15, frag[8..15]=K 24..31
__device__ __forceinline__ v16bf load_b_frag(const __bf16* __restrict__ Bw, int ldb) {
  const int lane = threadIdx.x & 31;
  const int c    = lane & 15;
  const int koff = (lane >> 4) << 3;   // 0 or 8
  const __bf16* p = Bw + (size_t)c * ldb + koff;
  v8bf lo = *(const v8bf*)p;
  v8bf hi = *(const v8bf*)(p + 16);
  v16bf f;
#pragma unroll
  for (int j = 0; j < 8; ++j) { f[j] = lo[j]; f[8 + j] = hi[j]; }
  return f;
}

// 8 WMMAs for a 32x64 tile step (2 row-halves x 4 col-groups).
__device__ __forceinline__ void wmma8(const v16bf& a0, const v16bf& a1,
                                      const v16bf& b0, const v16bf& b1,
                                      const v16bf& b2, const v16bf& b3,
                                      v8f acc[8]) {
  acc[0] = WMMA_BF16(a0, b0, acc[0]);
  acc[1] = WMMA_BF16(a0, b1, acc[1]);
  acc[2] = WMMA_BF16(a0, b2, acc[2]);
  acc[3] = WMMA_BF16(a0, b3, acc[3]);
  acc[4] = WMMA_BF16(a1, b0, acc[4]);
  acc[5] = WMMA_BF16(a1, b1, acc[5]);
  acc[6] = WMMA_BF16(a1, b2, acc[6]);
  acc[7] = WMMA_BF16(a1, b3, acc[7]);
}

// Dense NT 32x64-tile K loop: all 6 fragment loads hoisted before the WMMA burst.
__device__ __forceinline__ void gemm_32x64(const __bf16* __restrict__ A, int lda,
                                           const __bf16* __restrict__ Bm, int ldb,
                                           int K, v8f acc[8]) {
  for (int k0 = 0; k0 < K; k0 += 32) {
    v16bf a0 = load_a_frag(A + k0, lda);
    v16bf a1 = load_a_frag(A + (size_t)16 * lda + k0, lda);
    v16bf b0 = load_b_frag(Bm + k0, ldb);
    v16bf b1 = load_b_frag(Bm + (size_t)16 * ldb + k0, ldb);
    v16bf b2 = load_b_frag(Bm + (size_t)32 * ldb + k0, ldb);
    v16bf b3 = load_b_frag(Bm + (size_t)48 * ldb + k0, ldb);
    wmma8(a0, a1, b0, b1, b2, b3, acc);
  }
}

// ---------- elementwise conversion ----------

__global__ void cvt_f32_to_bf16(const float* __restrict__ in, __bf16* __restrict__ out, int n) {
  int i = blockIdx.x * blockDim.x + threadIdx.x;
  if (i < n) out[i] = (__bf16)in[i];
}

// ---------- build circulant filter kernel ----------
// w_ext[h][i] = w_h[i mod N],  i in [0,2N), where
// w_h[t] = (1/N) * sum_k atil_h[k] * cos(2*pi*k*t/N),
// atil_h[k] = 0.5*(a_h[k] + a_h[(N-k)%N]), a_h the reference's one-sided mask.
__global__ void build_filter(const float* __restrict__ freq, __bf16* __restrict__ wext) {
  int gid = blockIdx.x * blockDim.x + threadIdx.x;
  if (gid >= HEADS * 2 * SEQ) return;
  int h = gid / (2 * SEQ);
  int t = gid & (SEQ - 1);
  float sig    = 1.f / (1.f + __expf(-freq[h]));
  float cutoff = floorf((float)SEQ * sig);
  const float invdecay = 4.f / (float)SEQ;              // 1/(N/4)
  const float twopi_n  = 6.283185307179586f / (float)SEQ;
  float acc = 0.f;
  for (int k = 0; k < SEQ; ++k) {
    float r1 = (float)k - cutoff;
    float a1 = (r1 >= 0.f) ? __expf(-r1 * invdecay) : 1.f;
    int   km = (SEQ - k) & (SEQ - 1);
    float r2 = (float)km - cutoff;
    float a2 = (r2 >= 0.f) ? __expf(-r2 * invdecay) : 1.f;
    float at = 0.5f * (a1 + a2);
    int   ph = (k * t) & (SEQ - 1);                     // (k*t) mod N, exact
    acc += at * __cosf(twopi_n * (float)ph);
  }
  wext[gid] = (__bf16)(acc / (float)SEQ);
}

// ---------- GEMM1: qkv = x @ qkv_w.T, scatter into qT/kT/vT (B,H,D,N) bf16 ----------

__global__ void __launch_bounds__(256) qkv_gemm(
    const __bf16* __restrict__ xb, const __bf16* __restrict__ wb,
    __bf16* __restrict__ qT, __bf16* __restrict__ kT, __bf16* __restrict__ vT) {
  const int wid = (blockIdx.x * blockDim.x + threadIdx.x) >> 5;
  constexpr int TM = (BATCH * SEQ) / 32;  // 128
  constexpr int TN = QKV / 64;            // 36
  if (wid >= TM * TN) return;
  const int tm = wid % TM, tn = wid / TM;
  const int row0 = tm * 32, col0 = tn * 64;
  v8f acc[8] = {};
  gemm_32x64(xb + (size_t)row0 * DIM, DIM, wb + (size_t)col0 * DIM, DIM, DIM, acc);
  const int lane = threadIdx.x & 31;
  const int cN = lane & 15;
  const int rb = (lane >> 4) << 3;
#pragma unroll
  for (int g = 0; g < 2; ++g) {
#pragma unroll
    for (int i = 0; i < 4; ++i) {
      int col = col0 + i * 16 + cN;
      int s   = col / DIM;
      int rem = col - s * DIM;
      int h   = rem >> 6;
      int d   = rem & 63;
      __bf16* dst = (s == 0) ? qT : (s == 1) ? kT : vT;
#pragma unroll
      for (int r = 0; r < 8; ++r) {
        int row = row0 + g * 16 + rb + r;
        int b   = row >> 11;          // / SEQ
        int n   = row & (SEQ - 1);
        dst[((size_t)((b * HEADS + h) * HD + d)) * SEQ + n] = (__bf16)acc[g * 4 + i][r];
      }
    }
  }
}

// ---------- GEMM2: circulant filter; out[n,d] = sum_m w[(n-m)%N] * in[m,d] ----------
// A fragment gathered from w_ext (A[n][m] = w_ext[h][N + n - m]); B from (B,H,D,N).

__global__ void __launch_bounds__(256) filter_gemm(
    const __bf16* __restrict__ wext,
    const __bf16* __restrict__ qT, const __bf16* __restrict__ kT,
    __bf16* __restrict__ qf, __bf16* __restrict__ kf) {
  const int wid = (blockIdx.x * blockDim.x + threadIdx.x) >> 5;
  constexpr int TM  = SEQ / 32;       // 64
  constexpr int TOT = 2 * BH * TM;    // 3072
  if (wid >= TOT) return;
  const int tm = wid % TM;
  const int bh = (wid / TM) % BH;
  const int s  = wid / (TM * BH);
  const int h  = bh % HEADS;
  const int n0 = tm * 32;
  const __bf16* src  = ((s == 0) ? qT : kT) + (size_t)bh * HD * SEQ;
  __bf16*       dstF = ((s == 0) ? qf : kf) + (size_t)bh * SEQ * HD;
  const __bf16* wrow = wext + (size_t)h * (2 * SEQ);
  const int lane  = threadIdx.x & 31;
  const int m     = lane & 15;
  const int ko8   = (lane >> 4) << 3;
  const int nrow0 = n0 + m;
  const int nrow1 = n0 + 16 + m;
  v8f acc[8] = {};
  for (int k0 = 0; k0 < SEQ; k0 += 32) {
    const __bf16* wp0 = wrow + (SEQ + nrow0 - k0 - ko8);
    const __bf16* wp1 = wrow + (SEQ + nrow1 - k0 - ko8);
    v16bf a0, a1;
#pragma unroll
    for (int j = 0; j < 8; ++j) {
      a0[j] = wp0[-j];  a0[8 + j] = wp0[-16 - j];
      a1[j] = wp1[-j];  a1[8 + j] = wp1[-16 - j];
    }
    v16bf b0 = load_b_frag(src + k0, SEQ);
    v16bf b1 = load_b_frag(src + (size_t)16 * SEQ + k0, SEQ);
    v16bf b2 = load_b_frag(src + (size_t)32 * SEQ + k0, SEQ);
    v16bf b3 = load_b_frag(src + (size_t)48 * SEQ + k0, SEQ);
    wmma8(a0, a1, b0, b1, b2, b3, acc);
  }
  const int cN = lane & 15;
  const int rb = (lane >> 4) << 3;
#pragma unroll
  for (int g = 0; g < 2; ++g) {
#pragma unroll
    for (int i = 0; i < 4; ++i) {
      int d = i * 16 + cN;
#pragma unroll
      for (int r = 0; r < 8; ++r) {
        int n = n0 + g * 16 + rb + r;
        dstF[(size_t)n * HD + d] = (__bf16)acc[g * 4 + i][r];
      }
    }
  }
}

// ---------- GEMM3: S = scale * q_filt @ k_filt^T ----------

__global__ void __launch_bounds__(256) scores_gemm(
    const __bf16* __restrict__ qf, const __bf16* __restrict__ kf,
    __bf16* __restrict__ S) {
  const int wid = (blockIdx.x * blockDim.x + threadIdx.x) >> 5;
  constexpr int TM = SEQ / 32;   // 64
  constexpr int TN = SEQ / 64;   // 32
  constexpr int TOT = BH * TM * TN;
  if (wid >= TOT) return;
  const int tm = wid % TM;
  const int tn = (wid / TM) % TN;
  const int bh = wid / (TM * TN);
  const int row0 = tm * 32, col0 = tn * 64;
  v8f acc[8] = {};
  gemm_32x64(qf + (size_t)bh * SEQ * HD + (size_t)row0 * HD, HD,
             kf + (size_t)bh * SEQ * HD + (size_t)col0 * HD, HD, HD, acc);
  const float scale = 0.125f;   // HD^-0.5
  const int lane = threadIdx.x & 31;
  const int cN = lane & 15;
  const int rb = (lane >> 4) << 3;
  __bf16* Sb = S + (size_t)bh * SEQ * SEQ;
#pragma unroll
  for (int g = 0; g < 2; ++g) {
#pragma unroll
    for (int i = 0; i < 4; ++i) {
      int col = col0 + i * 16 + cN;
#pragma unroll
      for (int r = 0; r < 8; ++r) {
        int row = row0 + g * 16 + rb + r;
        Sb[(size_t)row * SEQ + col] = (__bf16)(acc[g * 4 + i][r] * scale);
      }
    }
  }
}

// ---------- softmax over rows of S (in place, bf16) ----------

__global__ void __launch_bounds__(256) softmax_rows(__bf16* __restrict__ S) {
  __shared__ float red[256];
  const int t = threadIdx.x;
  __bf16* row = S + (size_t)blockIdx.x * SEQ;
  float v[8];
  float mx = -3.4e38f;
#pragma unroll
  for (int i = 0; i < 8; ++i) {
    v[i] = (float)row[t * 8 + i];
    mx = fmaxf(mx, v[i]);
  }
  red[t] = mx; __syncthreads();
  for (int s = 128; s > 0; s >>= 1) { if (t < s) red[t] = fmaxf(red[t], red[t + s]); __syncthreads(); }
  mx = red[0]; __syncthreads();
  float sum = 0.f;
#pragma unroll
  for (int i = 0; i < 8; ++i) { v[i] = __expf(v[i] - mx); sum += v[i]; }
  red[t] = sum; __syncthreads();
  for (int s = 128; s > 0; s >>= 1) { if (t < s) red[t] += red[t + s]; __syncthreads(); }
  float inv = 1.f / red[0];
#pragma unroll
  for (int i = 0; i < 8; ++i) row[t * 8 + i] = (__bf16)(v[i] * inv);
}

// ---------- GEMM4: y = P @ v ; written as (B,N,C) bf16 ----------

__global__ void __launch_bounds__(256) attnv_gemm(
    const __bf16* __restrict__ P, const __bf16* __restrict__ vT,
    __bf16* __restrict__ y) {
  const int wid = (blockIdx.x * blockDim.x + threadIdx.x) >> 5;
  constexpr int TM = SEQ / 32;     // 64
  constexpr int TOT = BH * TM;     // 1536
  if (wid >= TOT) return;
  const int tm = wid % TM;
  const int bh = wid / TM;
  const int b = bh / HEADS, h = bh % HEADS;
  const int row0 = tm * 32;
  v8f acc[8] = {};
  gemm_32x64(P + (size_t)bh * SEQ * SEQ + (size_t)row0 * SEQ, SEQ,
             vT + (size_t)bh * HD * SEQ, SEQ, SEQ, acc);
  const int lane = threadIdx.x & 31;
  const int cN = lane & 15;
  const int rb = (lane >> 4) << 3;
#pragma unroll
  for (int g = 0; g < 2; ++g) {
#pragma unroll
    for (int i = 0; i < 4; ++i) {
      int d = i * 16 + cN;
#pragma unroll
      for (int r = 0; r < 8; ++r) {
        int n = row0 + g * 16 + rb + r;
        y[((size_t)(b * SEQ + n)) * DIM + h * HD + d] = (__bf16)acc[g * 4 + i][r];
      }
    }
  }
}

// ---------- GEMM5: out = y @ proj_w.T + proj_b (f32 output) ----------

__global__ void __launch_bounds__(256) proj_gemm(
    const __bf16* __restrict__ y, const __bf16* __restrict__ wb,
    const float* __restrict__ bias, float* __restrict__ out) {
  const int wid = (blockIdx.x * blockDim.x + threadIdx.x) >> 5;
  constexpr int TM = (BATCH * SEQ) / 32;  // 128
  constexpr int TN = DIM / 64;            // 12
  if (wid >= TM * TN) return;
  const int tm = wid % TM, tn = wid / TM;
  const int row0 = tm * 32, col0 = tn * 64;
  v8f acc[8] = {};
  gemm_32x64(y + (size_t)row0 * DIM, DIM, wb + (size_t)col0 * DIM, DIM, DIM, acc);
  const int lane = threadIdx.x & 31;
  const int cN = lane & 15;
  const int rb = (lane >> 4) << 3;
#pragma unroll
  for (int g = 0; g < 2; ++g) {
#pragma unroll
    for (int i = 0; i < 4; ++i) {
      int col = col0 + i * 16 + cN;
      float bsv = bias[col];
#pragma unroll
      for (int r = 0; r < 8; ++r) {
        int row = row0 + g * 16 + rb + r;
        out[(size_t)row * DIM + col] = acc[g * 4 + i][r] + bsv;
      }
    }
  }
}

// ---------- host launch ----------

extern "C" void kernel_launch(void* const* d_in, const int* in_sizes, int n_in,
                              void* d_out, int out_size, void* d_ws, size_t ws_size,
                              hipStream_t stream) {
  (void)in_sizes; (void)n_in; (void)out_size; (void)ws_size;
  const float* x      = (const float*)d_in[0];
  const float* qkv_w  = (const float*)d_in[1];
  const float* proj_w = (const float*)d_in[2];
  const float* proj_b = (const float*)d_in[3];
  const float* freq   = (const float*)d_in[4];

  char* ws = (char*)d_ws;
  size_t off = 0;
  auto take = [&](size_t elems) -> __bf16* {
    __bf16* p = (__bf16*)(ws + off);
    off += (elems * sizeof(__bf16) + 255) & ~(size_t)255;
    return p;
  };
  __bf16* xb   = take((size_t)BATCH * SEQ * DIM);
  __bf16* wqb  = take((size_t)QKV * DIM);
  __bf16* wpb  = take((size_t)DIM * DIM);
  __bf16* wext = take((size_t)HEADS * 2 * SEQ);
  __bf16* qT   = take((size_t)BH * HD * SEQ);
  __bf16* kT   = take((size_t)BH * HD * SEQ);
  __bf16* vT   = take((size_t)BH * HD * SEQ);
  __bf16* qf   = take((size_t)BH * SEQ * HD);
  __bf16* kf   = take((size_t)BH * SEQ * HD);
  __bf16* y    = take((size_t)BATCH * SEQ * DIM);
  __bf16* S    = take((size_t)BH * SEQ * SEQ);

  auto cdiv = [](int a, int b) { return (a + b - 1) / b; };

  cvt_f32_to_bf16<<<cdiv(BATCH * SEQ * DIM, 256), 256, 0, stream>>>(x, xb, BATCH * SEQ * DIM);
  cvt_f32_to_bf16<<<cdiv(QKV * DIM, 256), 256, 0, stream>>>(qkv_w, wqb, QKV * DIM);
  cvt_f32_to_bf16<<<cdiv(DIM * DIM, 256), 256, 0, stream>>>(proj_w, wpb, DIM * DIM);
  build_filter<<<cdiv(HEADS * 2 * SEQ, 256), 256, 0, stream>>>(freq, wext);

  qkv_gemm   <<<cdiv(((BATCH * SEQ) / 32) * (QKV / 64), 8), 256, 0, stream>>>(xb, wqb, qT, kT, vT);
  filter_gemm<<<cdiv(2 * BH * (SEQ / 32), 8),            256, 0, stream>>>(wext, qT, kT, qf, kf);
  scores_gemm<<<cdiv(BH * (SEQ / 32) * (SEQ / 64), 8),   256, 0, stream>>>(qf, kf, S);
  softmax_rows<<<BH * SEQ, 256, 0, stream>>>(S);
  attnv_gemm <<<cdiv(BH * (SEQ / 32), 8),                256, 0, stream>>>(S, vT, y);
  proj_gemm  <<<cdiv(((BATCH * SEQ) / 32) * (DIM / 64), 8), 256, 0, stream>>>(y, wpb, proj_b, (float*)d_out);
}
